// SelfAttentionNarrow_86079734547081
// MI455X (gfx1250) — matrix operand order repaired and checked by reading, not compiled, not run on hardware
//
#include <hip/hip_runtime.h>
#include <hip/hip_bf16.h>

typedef __attribute__((ext_vector_type(16))) __bf16 v16bf;
typedef __attribute__((ext_vector_type(8)))  __bf16 v8bf;
typedef __attribute__((ext_vector_type(8)))  float  v8f;

#define BATCH 4
#define SEQ   2048
#define HEADS 8
#define HDIM  96
#define KDIM  768

__device__ __forceinline__ v8f wmma_bf16(v16bf a, v16bf b, v8f c) {
  // D = A(16x32 bf16) * B(32x16 bf16) + C(16x16 f32)
  return __builtin_amdgcn_wmma_f32_16x16x32_bf16(
      /*neg_a=*/false, a, /*neg_b=*/false, b,
      /*c_mod=*/(short)0, c, /*reuse_a=*/false, /*reuse_b=*/false);
}

__device__ __forceinline__ v16bf cat8(v8bf lo, v8bf hi) {
  return __builtin_shufflevector(lo, hi, 0,1,2,3,4,5,6,7,8,9,10,11,12,13,14,15);
}

// ---------------------------------------------------------------------------
// Kernel 0: convert weights to bf16 once (fold softmax scale 1/sqrt(768) into Wq)
// ---------------------------------------------------------------------------
__global__ __launch_bounds__(256)
void wconv_kernel(const float* __restrict__ Wu, const float* __restrict__ Wq,
                  const float* __restrict__ Wk, const float* __restrict__ Wv,
                  __bf16* __restrict__ wub, __bf16* __restrict__ wqkv) {
  int i = blockIdx.x * 256 + threadIdx.x;
  if (i < KDIM * KDIM) wub[i] = (__bf16)Wu[i];
  int j = i - KDIM * KDIM;
  if (j >= 0 && j < 3 * HDIM * HDIM) {
    int pj = j / (HDIM * HDIM);
    int r  = j % (HDIM * HDIM);
    const float inv = 0.036084391824351613f; // 1/sqrt(768) = (1/768^0.25)^2
    float v = (pj == 0) ? Wq[r] * inv : ((pj == 1) ? Wk[r] : Wv[r]);
    wqkv[j] = (__bf16)v;
  }
}

// ---------------------------------------------------------------------------
// Kernel 1: QKV projection.  One wave computes a 16(t) x 96(o) tile of q,k,v
// for one head.  q,k stored [b*H + h][t][d] row-major; v stored transposed
// [b*H + h][d][t] so attention's P*V B-fragments are contiguous.
// ---------------------------------------------------------------------------
__global__ __launch_bounds__(32)
void qkv_kernel(const float* __restrict__ x, const __bf16* __restrict__ wqkv,
                __bf16* __restrict__ qb, __bf16* __restrict__ kb,
                __bf16* __restrict__ vt) {
  const int lane = threadIdx.x;
  const int l15  = lane & 15;
  const int lh   = lane >> 4;
  const int hh   = blockIdx.y;
  const int row0 = blockIdx.x * 16;       // flattened b*t tile base
  const int bb   = row0 / SEQ;
  const int t0   = row0 % SEQ;

  // A fragments (16x32 bf16): lane = (khalf<<4)|M; elem e -> K = e + 16*khalf
  v16bf xa[3];
  const float* xr = x + (size_t)(row0 + l15) * KDIM + hh * HDIM;
#pragma unroll
  for (int kk = 0; kk < 3; ++kk) {
    const float* p = xr + kk * 32 + 16 * lh;
    v8f lo = *(const v8f*)(p);
    v8f hi = *(const v8f*)(p + 8);
    v16bf a;
#pragma unroll
    for (int e = 0; e < 8; ++e) { a[e] = (__bf16)lo[e]; a[e + 8] = (__bf16)hi[e]; }
    xa[kk] = a;
  }

#pragma unroll
  for (int pj = 0; pj < 3; ++pj) {
    const __bf16* W = wqkv + pj * (HDIM * HDIM);
#pragma unroll
    for (int nb = 0; nb < 6; ++nb) {
      v8f c = {};
#pragma unroll
      for (int kk = 0; kk < 3; ++kk) {
        // B fragment (32x16): lane = (khalf<<4)|N; elem e -> K = e + 16*khalf
        const __bf16* wp = W + (size_t)(nb * 16 + l15) * HDIM + kk * 32 + 16 * lh;
        v16bf bw = *(const v16bf*)wp;
        c = wmma_bf16(xa[kk], bw, c);
      }
      // C layout: lane = (mhalf<<4)|N; elem r -> M = r + 8*mhalf
      if (pj < 2) {
        __bf16* dst = (pj == 0 ? qb : kb) + ((size_t)(bb * HEADS + hh) * SEQ) * HDIM;
#pragma unroll
        for (int r = 0; r < 8; ++r)
          dst[(size_t)(t0 + r + 8 * lh) * HDIM + nb * 16 + l15] = (__bf16)c[r];
      } else {
        __bf16* dst = vt + ((size_t)(bb * HEADS + hh) * HDIM + nb * 16 + l15) * SEQ;
#pragma unroll
        for (int r = 0; r < 8; ++r)
          dst[t0 + r + 8 * lh] = (__bf16)c[r];
      }
    }
  }
}

// ---------------------------------------------------------------------------
// Kernel 2: flash attention.  One wave owns 16 query rows of one (b,h),
// streams keys/values in 64-wide tiles with online softmax.  Row sums are
// computed with a WMMA against an all-ones B matrix (lands in the same C
// slots as lrun), so only the row-max needs cross-lane shuffles.
// ---------------------------------------------------------------------------
__global__ __launch_bounds__(32)
void attn_kernel(const __bf16* __restrict__ qb, const __bf16* __restrict__ kbuf,
                 const __bf16* __restrict__ vt, __bf16* __restrict__ ao) {
  __shared__ __bf16 Pl[16 * 64];
  const int lane = threadIdx.x;
  const int l15  = lane & 15;
  const int lh   = lane >> 4;
  const int bh   = blockIdx.y;          // b*HEADS + h
  const int bb   = bh / HEADS;
  const int hh   = bh % HEADS;
  const int tq0  = blockIdx.x * 16;

  // Preload Q A-fragments for the 16 query rows (scale already folded into Wq)
  v16bf qa[3];
  const __bf16* qrow = qb + ((size_t)bh * SEQ + tq0) * HDIM;
#pragma unroll
  for (int kk = 0; kk < 3; ++kk) {
    const __bf16* p = qrow + (size_t)l15 * HDIM + kk * 32 + 16 * lh;
    v8bf lo = *(const v8bf*)(p);
    v8bf hi = *(const v8bf*)(p + 8);
    qa[kk] = cat8(lo, hi);
  }

  // all-ones B fragment for row-sum WMMA
  v16bf onesb;
#pragma unroll
  for (int e = 0; e < 16; ++e) onesb[e] = (__bf16)1.0f;

  const __bf16* krows = kbuf + (size_t)bh * SEQ * HDIM;
  const __bf16* vrows = vt + (size_t)bh * HDIM * SEQ;

  float mrun[8], lrun[8];
  v8f acc[6] = {};
#pragma unroll
  for (int r = 0; r < 8; ++r) { mrun[r] = -1.0e30f; lrun[r] = 0.0f; }

  for (int s0 = 0; s0 < SEQ; s0 += 64) {
    if (s0 + 64 < SEQ) {  // stream next K/V tiles toward L2/L0
      __builtin_prefetch(krows + (size_t)(s0 + 64 + l15) * HDIM, 0, 1);
      __builtin_prefetch(krows + (size_t)(s0 + 96 + l15) * HDIM, 0, 1);
      __builtin_prefetch(vrows + (size_t)l15 * SEQ + s0 + 64, 0, 1);
    }
    // ---- scores: four 16x16 tiles over d=96 (3 K-steps each) ----
    v8f sc[4] = {};
#pragma unroll
    for (int j = 0; j < 4; ++j)
#pragma unroll
      for (int kk = 0; kk < 3; ++kk) {
        const __bf16* p = krows + (size_t)(s0 + j * 16 + l15) * HDIM + kk * 32 + 16 * lh;
        sc[j] = wmma_bf16(qa[kk], *(const v16bf*)p, sc[j]);
      }
    // ---- online softmax: row-max via shuffles, P -> LDS as bf16 ----
    float al[8];
#pragma unroll
    for (int r = 0; r < 8; ++r) {
      float t = fmaxf(fmaxf(sc[0][r], sc[1][r]), fmaxf(sc[2][r], sc[3][r]));
      t = fmaxf(t, __shfl_xor(t, 1, 16));
      t = fmaxf(t, __shfl_xor(t, 2, 16));
      t = fmaxf(t, __shfl_xor(t, 4, 16));
      t = fmaxf(t, __shfl_xor(t, 8, 16));
      float mnew = fmaxf(mrun[r], t);
      al[r] = __expf(mrun[r] - mnew);
      mrun[r] = mnew;
#pragma unroll
      for (int j = 0; j < 4; ++j)
        Pl[(r + 8 * lh) * 64 + j * 16 + l15] = (__bf16)__expf(sc[j][r] - mnew);
    }
    __syncthreads();
    // ---- P as two A fragments (K = 0..31, 32..63 of this key tile) ----
    const __bf16* pp = &Pl[l15 * 64 + 16 * lh];
    v16bf pa0 = cat8(*(const v8bf*)(pp),      *(const v8bf*)(pp + 8));
    v16bf pa1 = cat8(*(const v8bf*)(pp + 32), *(const v8bf*)(pp + 40));
    // ---- row sums via WMMA against ones (replaces 32 shuffles) ----
    v8f rsum = {};
    rsum = wmma_bf16(pa0, onesb, rsum);
    rsum = wmma_bf16(pa1, onesb, rsum);
#pragma unroll
    for (int r = 0; r < 8; ++r) lrun[r] = lrun[r] * al[r] + rsum[r];
    // ---- rescale accumulator, then acc += P * V (K = 64 keys) ----
#pragma unroll
    for (int i = 0; i < 6; ++i)
#pragma unroll
      for (int r = 0; r < 8; ++r) acc[i][r] *= al[r];
#pragma unroll
    for (int nb = 0; nb < 6; ++nb) {
      const __bf16* vp = vrows + (size_t)(nb * 16 + l15) * SEQ + s0 + 16 * lh;
      acc[nb] = wmma_bf16(pa0, *(const v16bf*)vp, acc[nb]);
      acc[nb] = wmma_bf16(pa1, *(const v16bf*)(vp + 32), acc[nb]);
    }
    __syncthreads();
  }

  // ---- normalize and emit attention output [b][t][h*96+d] as bf16 ----
#pragma unroll
  for (int nb = 0; nb < 6; ++nb)
#pragma unroll
    for (int r = 0; r < 8; ++r) {
      int trow = tq0 + r + 8 * lh;
      float v = acc[nb][r] / lrun[r];
      ao[((size_t)bb * SEQ + trow) * KDIM + hh * HDIM + nb * 16 + l15] = (__bf16)v;
    }
}

// ---------------------------------------------------------------------------
// Kernel 3: output projection  Y = AO @ Wu^T + bu   (f32 out)
// One wave computes a 16(t) x 64(o) tile; K-loop of 24 steps of 32.
// ---------------------------------------------------------------------------
__global__ __launch_bounds__(32)
void proj_kernel(const __bf16* __restrict__ ao, const __bf16* __restrict__ wub,
                 const float* __restrict__ bu, float* __restrict__ out) {
  const int lane = threadIdx.x;
  const int l15  = lane & 15;
  const int lh   = lane >> 4;
  const int row0 = blockIdx.x * 16;
  const int ob0  = blockIdx.y * 64;

  v8f c[4] = {};
  for (int cb = 0; cb < KDIM; cb += 32) {
    const __bf16* ap = ao + (size_t)(row0 + l15) * KDIM + cb + 16 * lh;
    v16bf a = cat8(*(const v8bf*)(ap), *(const v8bf*)(ap + 8));
#pragma unroll
    for (int j = 0; j < 4; ++j) {
      const __bf16* wp = wub + (size_t)(ob0 + j * 16 + l15) * KDIM + cb + 16 * lh;
      c[j] = wmma_bf16(a, *(const v16bf*)wp, c[j]);
    }
  }
#pragma unroll
  for (int j = 0; j < 4; ++j)
#pragma unroll
    for (int r = 0; r < 8; ++r) {
      int o = ob0 + j * 16 + l15;
      out[(size_t)(row0 + r + 8 * lh) * KDIM + o] = c[j][r] + bu[o];
    }
}

// ---------------------------------------------------------------------------
extern "C" void kernel_launch(void* const* d_in, const int* in_sizes, int n_in,
                              void* d_out, int out_size, void* d_ws, size_t ws_size,
                              hipStream_t stream) {
  (void)in_sizes; (void)n_in; (void)out_size; (void)ws_size;
  const float* x  = (const float*)d_in[0];
  const float* Wq = (const float*)d_in[1];
  const float* Wk = (const float*)d_in[2];
  const float* Wv = (const float*)d_in[3];
  const float* Wu = (const float*)d_in[4];
  const float* bu = (const float*)d_in[5];

  char* ws = (char*)d_ws;
  const size_t S1 = (size_t)BATCH * HEADS * SEQ * HDIM * 2;  // 12.6 MB each
  __bf16* qb   = (__bf16*)(ws);
  __bf16* kb   = (__bf16*)(ws + S1);
  __bf16* vt   = (__bf16*)(ws + 2 * S1);
  __bf16* ao   = (__bf16*)(ws + 3 * S1);
  __bf16* wub  = (__bf16*)(ws + 4 * S1);
  __bf16* wqkv = (__bf16*)(ws + 4 * S1 + (size_t)KDIM * KDIM * 2);

  const int nconv = KDIM * KDIM + 3 * HDIM * HDIM;
  wconv_kernel<<<dim3((nconv + 255) / 256), 256, 0, stream>>>(Wu, Wq, Wk, Wv, wub, wqkv);
  qkv_kernel<<<dim3((BATCH * SEQ) / 16, HEADS), 32, 0, stream>>>(x, wqkv, qb, kb, vt);
  attn_kernel<<<dim3(SEQ / 16, BATCH * HEADS), 32, 0, stream>>>(qb, kb, vt, ao);
  proj_kernel<<<dim3((BATCH * SEQ) / 16, KDIM / 64), 32, 0, stream>>>(ao, wub, bu, (float*)d_out);
}
